// uiAttention_67130338836549
// MI455X (gfx1250) — matrix operand order repaired and checked by reading, not compiled, not run on hardware
//
#include <hip/hip_runtime.h>
#include <hip/hip_bf16.h>

// MI455X / gfx1250: wave32, WMMA bf16 16x16x32 with f32 accumulate.
// One block per m (4096 blocks), 256 threads = 8 waves; wave w owns output
// feature tile [16w, 16w+16). Memory-bound target: ~1.05 GB streamed once
// (~45 us at 23.3 TB/s); bf16 WMMA keeps compute well under that floor.

typedef __attribute__((ext_vector_type(16))) __bf16 bf16x16;
typedef __attribute__((ext_vector_type(8)))  float  f32x8;

#define XS 72   // padded LDS k-stride in shorts (144 B, avoids 64-bank conflicts)

__device__ __forceinline__ unsigned short f2bf(float f) {
    unsigned int u = __float_as_uint(f);
    unsigned int r = u + 0x7FFFu + ((u >> 16) & 1u);   // round-to-nearest-even
    return (unsigned short)(r >> 16);
}

union Frag {
    bf16x16 v;
    uint4   q[2];
};

// A 16x32 bf16 (ISA 7.12.2): lane-half g: e=0..7 -> k=8g+e ; e=8..15 -> k=16+8g+(e-8)
__device__ __forceinline__ void loadFragA(Frag& f, const unsigned short* p) {
    f.q[0] = *(const uint4*)(p);        // k = base .. base+7
    f.q[1] = *(const uint4*)(p + 16);   // k = base+16 .. base+23
}
// B 32x16 bf16: lane-half g: e=0..7 -> k=8g+e ; e=8..15 -> k=16+8g+(e-8)
__device__ __forceinline__ void loadFragB(Frag& f, const unsigned short* p) {
    f.q[0] = *(const uint4*)(p);
    f.q[1] = *(const uint4*)(p + 16);
}

__global__ __launch_bounds__(256, 1) void uiattn_wmma_kernel(
    const float* __restrict__ pus, const float* __restrict__ pis,
    const float* __restrict__ pss, const float* __restrict__ Cc,
    const float* __restrict__ W1,  const float* __restrict__ b1,
    const float* __restrict__ W2,
    const float* __restrict__ WR,  const float* __restrict__ bR,
    const float* __restrict__ Wo,  const float* __restrict__ bo,
    float* __restrict__ out)
{
    __shared__ unsigned short sX[112 * XS];   // activation chunk, bf16
    __shared__ unsigned short sW[128 * XS];   // weight chunk, transposed [col][k], bf16
    __shared__ float sScore[112];
    __shared__ float sWgt[112];
    __shared__ float sPool[128];

    const int m    = blockIdx.x;
    const int tid  = threadIdx.x;
    const int wv   = tid >> 5;        // wave id == output column tile
    const int lane = tid & 31;
    const int l15  = lane & 15;
    const int g    = lane >> 4;       // lane half
    const int e    = wv * 16 + l15;   // feature column owned in C/D fragments

    if (tid < 112) sScore[tid] = 0.f;

    f32x8 acc[7];
    #pragma unroll
    for (int t = 0; t < 7; ++t)
        #pragma unroll
        for (int j = 0; j < 8; ++j) acc[t][j] = 0.f;

    // ---------------- layer 1: h = relu([pus|pis|pss] @ W1 + b1) ----------------
    for (int c = 0; c < 6; ++c) {
        const float* src = (c < 2) ? pus : (c < 4) ? pis : pss;
        const int ko = (c & 1) * 64;
        // stage activation chunk (rows >= 100 zero-padded), f32 -> bf16
        for (int i = tid; i < 112 * 64; i += 256) {
            const int r = i >> 6, kk = i & 63;
            float v = (r < 100) ? src[((size_t)m * 100 + r) * 128 + ko + kk] : 0.f;
            sX[r * XS + kk] = f2bf(v);
        }
        // stage W1 chunk transposed: sW[col][k]
        for (int i = tid; i < 128 * 64; i += 256) {
            const int col = i & 127, kk = i >> 7;
            sW[col * XS + kk] = f2bf(W1[(size_t)(c * 64 + kk) * 128 + col]);
        }
        __syncthreads();
        #pragma unroll
        for (int f = 0; f < 2; ++f) {
            // Preload B + all 7 A fragments, then issue WMMAs back-to-back so
            // LDS latency is batched under one drain instead of per-WMMA waits.
            Frag bfr; loadFragB(bfr, &sW[(wv * 16 + l15) * XS + f * 32 + 8 * g]);
            Frag afr[7];
            #pragma unroll
            for (int t = 0; t < 7; ++t)
                loadFragA(afr[t], &sX[(t * 16 + l15) * XS + f * 32 + 8 * g]);
            #pragma unroll
            for (int t = 0; t < 7; ++t)
                acc[t] = __builtin_amdgcn_wmma_f32_16x16x32_bf16(
                    false, afr[t].v, false, bfr.v, (short)0, acc[t], false, false);
        }
        __syncthreads();
    }

    // ---------------- scores = relu(h) . W2 (b2 cancels in softmax) -------------
    {
        const float be  = b1[e];
        const float w2e = W2[e];
        #pragma unroll
        for (int t = 0; t < 7; ++t) {
            #pragma unroll
            for (int j = 0; j < 8; ++j) {
                float p = fmaxf(acc[t][j] + be, 0.f) * w2e;
                p += __shfl_xor(p, 1);
                p += __shfl_xor(p, 2);
                p += __shfl_xor(p, 4);
                p += __shfl_xor(p, 8);
                if (l15 == 0) atomicAdd(&sScore[t * 16 + 8 * g + j], p);
            }
        }
    }
    __syncthreads();

    // ---------------- softmax over 100 paths (wave 0) ----------------------------
    if (tid < 32) {
        float mx = -3.4e38f;
        for (int n = lane; n < 100; n += 32) mx = fmaxf(mx, sScore[n]);
        #pragma unroll
        for (int o = 16; o; o >>= 1) mx = fmaxf(mx, __shfl_xor(mx, o));
        float sum = 0.f;
        for (int n = lane; n < 100; n += 32) {
            float ex = __expf(sScore[n] - mx);
            sWgt[n] = ex;
            sum += ex;
        }
        #pragma unroll
        for (int o = 16; o; o >>= 1) sum += __shfl_xor(sum, o);
        const float inv = 1.f / sum;
        for (int n = lane; n < 112; n += 32) sWgt[n] = (n < 100) ? sWgt[n] * inv : 0.f;
    }
    __syncthreads();

    // ---------------- layer 2: v = relu([C|pss] @ WR + bR) ----------------------
    #pragma unroll
    for (int t = 0; t < 7; ++t)
        #pragma unroll
        for (int j = 0; j < 8; ++j) acc[t][j] = 0.f;

    for (int c = 0; c < 4; ++c) {
        const float* src = (c < 2) ? Cc : pss;
        const int ko = (c & 1) * 64;
        for (int i = tid; i < 112 * 64; i += 256) {
            const int r = i >> 6, kk = i & 63;
            float v = (r < 100) ? src[((size_t)m * 100 + r) * 128 + ko + kk] : 0.f;
            sX[r * XS + kk] = f2bf(v);
        }
        for (int i = tid; i < 128 * 64; i += 256) {
            const int col = i & 127, kk = i >> 7;
            sW[col * XS + kk] = f2bf(WR[(size_t)(c * 64 + kk) * 128 + col]);
        }
        __syncthreads();
        #pragma unroll
        for (int f = 0; f < 2; ++f) {
            Frag bfr; loadFragB(bfr, &sW[(wv * 16 + l15) * XS + f * 32 + 8 * g]);
            Frag afr[7];
            #pragma unroll
            for (int t = 0; t < 7; ++t)
                loadFragA(afr[t], &sX[(t * 16 + l15) * XS + f * 32 + 8 * g]);
            #pragma unroll
            for (int t = 0; t < 7; ++t)
                acc[t] = __builtin_amdgcn_wmma_f32_16x16x32_bf16(
                    false, afr[t].v, false, bfr.v, (short)0, acc[t], false, false);
        }
        __syncthreads();
    }

    // ---------------- weighted pooling: pooled = softmax_w . v -------------------
    float pe = 0.f;
    {
        const float be = bR[e];
        #pragma unroll
        for (int t = 0; t < 7; ++t)
            #pragma unroll
            for (int j = 0; j < 8; ++j)
                pe += sWgt[t * 16 + 8 * g + j] * fmaxf(acc[t][j] + be, 0.f);
    }
    pe += __shfl_xor(pe, 16);     // combine the two lane halves of feature e
    if (g == 0) sPool[e] = pe;
    __syncthreads();

    // ---------------- lout: out = relu(pooled @ Wo + bo) -------------------------
    if (tid < 128) {
        float a = bo[tid];
        #pragma unroll 8
        for (int k = 0; k < 128; ++k) a += sPool[k] * Wo[(size_t)k * 128 + tid];
        out[(size_t)m * 128 + tid] = fmaxf(a, 0.f);
    }
}

extern "C" void kernel_launch(void* const* d_in, const int* in_sizes, int n_in,
                              void* d_out, int out_size, void* d_ws, size_t ws_size,
                              hipStream_t stream) {
    (void)in_sizes; (void)n_in; (void)out_size; (void)d_ws; (void)ws_size;
    // setup_inputs order: user(0) item(1) pus(2) pis(3) pss(4) C(5)
    //                     W1(6) b1(7) W2(8) b2(9) WR(10) bR(11) Wo(12) bo(13)
    const float* pus = (const float*)d_in[2];
    const float* pis = (const float*)d_in[3];
    const float* pss = (const float*)d_in[4];
    const float* Cc  = (const float*)d_in[5];
    const float* W1  = (const float*)d_in[6];
    const float* b1  = (const float*)d_in[7];
    const float* W2  = (const float*)d_in[8];
    const float* WR  = (const float*)d_in[10];
    const float* bR  = (const float*)d_in[11];
    const float* Wo  = (const float*)d_in[12];
    const float* bo  = (const float*)d_in[13];
    float* out = (float*)d_out;

    uiattn_wmma_kernel<<<dim3(4096), dim3(256), 0, stream>>>(
        pus, pis, pss, Cc, W1, b1, W2, WR, bR, Wo, bo, out);
}
